// AttentionBlock_66700842107369
// MI455X (gfx1250) — matrix operand order repaired and checked by reading, not compiled, not run on hardware
//
#include <hip/hip_runtime.h>
#include <hip/hip_bf16.h>

// ---------------------------------------------------------------------------
// CDNA5 (gfx1250) attention block:  out = Proj(Softmax(mask(RoPE(Q)RoPE(K)^T))V) + b
// wave32, WMMA f32_16x16x32_f16, f32 accumulation, TDM tile load for K.
// ---------------------------------------------------------------------------

typedef __attribute__((ext_vector_type(16))) _Float16     v16h;
typedef __attribute__((ext_vector_type(8)))  float        v8f;
typedef __attribute__((ext_vector_type(4)))  unsigned int v4u;
typedef __attribute__((ext_vector_type(8)))  int          v8i;
typedef __attribute__((ext_vector_type(4)))  int          v4i;

union F16Frag { v16h v; v4u u[2]; };

#if defined(__has_builtin)
#if __has_builtin(__builtin_amdgcn_tensor_load_to_lds) && __has_builtin(__builtin_amdgcn_s_wait_tensorcnt)
#define USE_TDM 1
#endif
#endif
#ifndef USE_TDM
#define USE_TDM 0
#endif

__device__ __forceinline__ v8f wmma_f16(v16h a, v16h b, v8f c) {
    // D = A(16x32 f16) * B(32x16 f16) + C(16x16 f32)
    return __builtin_amdgcn_wmma_f32_16x16x32_f16(
        /*neg_a=*/false, a, /*neg_b=*/false, b,
        /*c_mod=*/(short)0, c, /*reuse_a=*/false, /*reuse_b=*/false);
}

// A-matrix 16x32 f16 fragment, per ISA 7.12.2 (row-major source, ld in halves).
__device__ __forceinline__ v16h frag_a16(const _Float16* base, int ld) {
    const int lane = threadIdx.x & 31;
    const int row  = lane & 15;
    const int k0   = (lane >> 4) << 3;      // 0 or 8
    const _Float16* p = base + row * ld;
    F16Frag f;
    f.u[0] = *(const v4u*)(p + k0);         // K = k0 .. k0+7
    f.u[1] = *(const v4u*)(p + 16 + k0);    // K = 16+k0 .. 16+k0+7
    return f.v;
}

// B-matrix 32x16 f16 fragment from a TRANSPOSED store S[n][k] (ld in halves).
__device__ __forceinline__ v16h frag_b16(const _Float16* baseT, int ld) {
    const int lane = threadIdx.x & 31;
    const int n  = lane & 15;
    const int kb = (lane >> 4) << 3;        // 0 or 8
    const _Float16* p = baseT + n * ld + kb;
    F16Frag f;
    f.u[0] = *(const v4u*)(p);              // K = kb .. kb+7
    f.u[1] = *(const v4u*)(p + 16);         // K = 16+kb .. 16+kb+7
    return f.v;
}

#if USE_TDM
// TDM 2D tile load: gsrc row-major f16 [rows x cols], row stride = stride (halves),
// into LDS at lds_dst (densely packed cols per row). D# per ISA cdna5 8.3/8.4.
__device__ __forceinline__ void tdm_load_tile_f16(_Float16* lds_dst, const _Float16* gsrc,
                                                  int rows, int cols, int stride) {
    unsigned int lds_off = (unsigned int)(size_t)(void*)lds_dst;     // LDS aperture: low 32 bits
    unsigned long long ga = (unsigned long long)(size_t)(const void*)gsrc;
    v4u g0;
    g0.x = 1u;                                            // count=1, user descriptor
    g0.y = lds_off;                                       // lds_addr [63:32]
    g0.z = (unsigned int)(ga & 0xffffffffu);              // global_addr [95:64]
    g0.w = (unsigned int)((ga >> 32) & 0x01ffffffu)       // global_addr [120:96]
         | (2u << 30);                                    // type=2 ("image")
    const unsigned td1 = 1u << 20;                        // huge tensor_dim1: no OOB clipping
    v8i g1;
    g1[0] = (int)(1u << 16);                              // wg_mask=0, data_size=1 (2B)
    g1[1] = (int)((unsigned)cols << 16);                  // tensor_dim0[15:0] @ bits 63:48
    g1[2] = (int)(((unsigned)cols >> 16) & 0xffffu);      // tensor_dim0 hi; tensor_dim1 lo16=0
    g1[3] = (int)((td1 >> 16) & 0xffffu)                  // tensor_dim1 hi16
          | (int)((unsigned)cols << 16);                  // tile_dim0 @ bits 127:112
    g1[4] = (int)(unsigned)rows;                          // tile_dim1; tile_dim2=0
    g1[5] = (int)(unsigned)stride;                        // tensor_dim0_stride [191:160]
    g1[6] = 0;                                            // stride hi, dim1_stride lo
    g1[7] = 0;
    v4i z{};
#if __clang_major__ >= 23
    v8i z8{};
    __builtin_amdgcn_tensor_load_to_lds(g0, g1, z, z, z8, 0);
#else
    __builtin_amdgcn_tensor_load_to_lds(g0, g1, z, z, 0);
#endif
}
#endif

// ---------------------------------------------------------------------------
// f32 -> f16 cast
// ---------------------------------------------------------------------------
__global__ void cast_f32_to_f16(const float* __restrict__ in,
                                _Float16* __restrict__ out, size_t n) {
    size_t i = (size_t)blockIdx.x * blockDim.x + threadIdx.x;
    if (i < n) out[i] = (_Float16)in[i];
}

// ---------------------------------------------------------------------------
// GEMM: C[M][N] = A[M][K] * B[N][K]^T (+ bias). A,B f16 row-major, C f32.
// 128x128 block tile, 256 threads = 8 waves; double-buffered LDS; B-fragment
// batch loads so the 8 WMMAs issue back-to-back.
// ---------------------------------------------------------------------------
template <bool ADD_BIAS>
__global__ __launch_bounds__(256)
void gemm_f16_wmma(const _Float16* __restrict__ A, const _Float16* __restrict__ B,
                   const float* __restrict__ bias, float* __restrict__ C,
                   int M, int N, int K) {
    __shared__ _Float16 sA[2][128 * 32];
    __shared__ _Float16 sB[2][128 * 32];

    const int tid  = threadIdx.x;
    const int wave = tid >> 5;
    const int lane = tid & 31;
    const int m0 = blockIdx.y * 128;
    const int n0 = blockIdx.x * 128;

    // this thread's two 16B tile chunks (row r, col kc within 128x32 tile)
    const int c0 = tid,       r0 = c0 >> 2, kc0 = (c0 & 3) * 8;
    const int c1 = tid + 256, r1 = c1 >> 2, kc1 = (c1 & 3) * 8;

    v8f acc[8];
#pragma unroll
    for (int j = 0; j < 8; ++j) acc[j] = {};

    // prologue: fill buffer 0
    {
        *(v4u*)(sA[0] + r0 * 32 + kc0) = *(const v4u*)(A + (size_t)(m0 + r0) * K + kc0);
        *(v4u*)(sA[0] + r1 * 32 + kc1) = *(const v4u*)(A + (size_t)(m0 + r1) * K + kc1);
        *(v4u*)(sB[0] + r0 * 32 + kc0) = *(const v4u*)(B + (size_t)(n0 + r0) * K + kc0);
        *(v4u*)(sB[0] + r1 * 32 + kc1) = *(const v4u*)(B + (size_t)(n0 + r1) * K + kc1);
    }
    __syncthreads();

    int buf = 0;
    for (int kk = 0; kk < K; kk += 32) {
        const int nxt = buf ^ 1;
        if (kk + 32 < K) {
            // stage next K-slab into the other buffer (overlaps with WMMA below)
            const int kn = kk + 32;
            *(v4u*)(sA[nxt] + r0 * 32 + kc0) = *(const v4u*)(A + (size_t)(m0 + r0) * K + kn + kc0);
            *(v4u*)(sA[nxt] + r1 * 32 + kc1) = *(const v4u*)(A + (size_t)(m0 + r1) * K + kn + kc1);
            *(v4u*)(sB[nxt] + r0 * 32 + kc0) = *(const v4u*)(B + (size_t)(n0 + r0) * K + kn + kc0);
            *(v4u*)(sB[nxt] + r1 * 32 + kc1) = *(const v4u*)(B + (size_t)(n0 + r1) * K + kn + kc1);
            if (kk + 64 < K) {
                __builtin_prefetch(A + (size_t)(m0 + (tid >> 1)) * K + kk + 64, 0, 0);
                __builtin_prefetch(B + (size_t)(n0 + (tid >> 1)) * K + kk + 64, 0, 0);
            }
        }

        // batch all fragment loads, then an uninterrupted 8-WMMA chain
        v16h a = frag_a16(sA[buf] + wave * 16 * 32, 32);
        v16h bfr[8];
#pragma unroll
        for (int j = 0; j < 8; ++j) bfr[j] = frag_b16(sB[buf] + j * 16 * 32, 32);
#pragma unroll
        for (int j = 0; j < 8; ++j) acc[j] = wmma_f16(a, bfr[j], acc[j]);

        __syncthreads();   // reads of buf done; stores to nxt visible
        buf = nxt;
    }

    // C layout: lane holds col n0+16j+(lane&15); VGPR r -> row m0+16w+8*(lane>>4)+r
    const int nl = lane & 15;
    const int mh = (lane >> 4) * 8;
#pragma unroll
    for (int j = 0; j < 8; ++j) {
        int n = n0 + j * 16 + nl;
        float bv = ADD_BIAS ? bias[n] : 0.0f;
#pragma unroll
        for (int r = 0; r < 8; ++r) {
            int m = m0 + wave * 16 + mh + r;
            C[(size_t)m * N + n] = acc[j][r] + bv;
        }
    }
}

// ---------------------------------------------------------------------------
// RoPE + split qkv[N][3*H*D] f32 -> Q,K,V f16 in [H][N][D]  (D=128 interleaved)
// ---------------------------------------------------------------------------
__global__ void rope_split_kernel(const float* __restrict__ qkv,
                                  _Float16* __restrict__ Qo,
                                  _Float16* __restrict__ Ko,
                                  _Float16* __restrict__ Vo,
                                  int N, int H) {
    size_t idx = (size_t)blockIdx.x * blockDim.x + threadIdx.x;
    int i = (int)(idx & 63);          // rope pair index, D/2 = 64
    int h = (int)((idx >> 6) & 15);
    int n = (int)(idx >> 10);         // H*64 = 1024 per row
    if (n >= N) return;

    const int C3 = 3 * H * 128;
    const float* row = qkv + (size_t)n * C3;

    // theta = n * time^(-2i/D), time=10000
    float theta = (float)n * __expf(-(float)(2 * i) * (9.210340371976184f / 128.0f));
    float sn, cs;
    __sincosf(theta, &sn, &cs);

    float q0 = row[h * 128 + 2 * i], q1 = row[h * 128 + 2 * i + 1];
    float k0 = row[H * 128 + h * 128 + 2 * i], k1 = row[H * 128 + h * 128 + 2 * i + 1];
    float v0 = row[2 * H * 128 + h * 128 + 2 * i], v1 = row[2 * H * 128 + h * 128 + 2 * i + 1];

    size_t ob = ((size_t)h * N + n) * 128 + 2 * i;
    Qo[ob]     = (_Float16)(q0 * cs - q1 * sn);
    Qo[ob + 1] = (_Float16)(q1 * cs + q0 * sn);
    Ko[ob]     = (_Float16)(k0 * cs - k1 * sn);
    Ko[ob + 1] = (_Float16)(k1 * cs + k0 * sn);
    Vo[ob]     = (_Float16)v0;
    Vo[ob + 1] = (_Float16)v1;
}

// ---------------------------------------------------------------------------
// Flash attention (causal, online softmax). One block = (head, 128 query rows).
// 8 waves x 16 rows; key blocks of 64; D = 128. K tile via TDM when available.
// ---------------------------------------------------------------------------
__global__ __launch_bounds__(256)
void flash_attn_kernel(const _Float16* __restrict__ Q,
                       const _Float16* __restrict__ Kg,
                       const _Float16* __restrict__ V,
                       _Float16* __restrict__ Oa, int N, int H) {
    __shared__ _Float16 sK[64 * 128];     // [key][d]
    __shared__ _Float16 sVt[128 * 64];    // [d][key]  (transposed at load)
    __shared__ _Float16 sP[8][16 * 64];   // per-wave P staging (C-layout -> A-layout)

    const int tid  = threadIdx.x;
    const int wave = tid >> 5;
    const int lane = tid & 31;
    const int h  = blockIdx.y;
    const int m0 = blockIdx.x * 128;
    const int qrow0 = m0 + wave * 16;
    const float scale = 0.08838834764831845f;   // 1/sqrt(128)

    // Q fragments (16 rows x 128 dims = 4 A-frags), straight from global
    const _Float16* qbase = Q + ((size_t)h * N + qrow0) * 128;
    v16h qf[4];
#pragma unroll
    for (int d = 0; d < 4; ++d) qf[d] = frag_a16(qbase + d * 32, 128);

    v8f o[8];
#pragma unroll
    for (int j = 0; j < 8; ++j) o[j] = {};
    float mi[8], li[8];
#pragma unroll
    for (int r = 0; r < 8; ++r) { mi[r] = -3.0e38f; li[r] = 0.0f; }

    const int nl = lane & 15;
    const int mh = (lane >> 4) * 8;
    const int nkb = (m0 >> 6) + 2;   // 64-key blocks covering keys <= m0+127

    for (int kb = 0; kb < nkb; ++kb) {
        const int kbase = kb * 64;
        __syncthreads();

#if USE_TDM
        // K tile (64x128 f16) via Tensor Data Mover; one wave issues, fences
        // with TENSORcnt, block barrier below publishes to all waves.
        if (wave == 0) {
            tdm_load_tile_f16(sK, Kg + ((size_t)h * N + kbase) * 128, 64, 128, 128);
            __builtin_amdgcn_s_wait_tensorcnt(0);
        }
#endif
        // V tile load + transpose scatter (and K fallback without TDM)
#pragma unroll
        for (int it = 0; it < 4; ++it) {
            int c  = tid + it * 256;       // 0..1023
            int r  = c >> 4;               // key row 0..63
            int kc = (c & 15) * 8;         // d 0..120
            const size_t gro = ((size_t)h * N + kbase + r) * 128 + kc;
#if !USE_TDM
            *(v4u*)(sK + r * 128 + kc) = *(const v4u*)(Kg + gro);
#endif
            v4u vv = *(const v4u*)(V + gro);
            const _Float16* vh = (const _Float16*)&vv;
#pragma unroll
            for (int e = 0; e < 8; ++e) sVt[(kc + e) * 64 + r] = vh[e];
        }
        if (kbase + 64 < N) {
            __builtin_prefetch(Kg + ((size_t)h * N + kbase + 64 + (tid >> 2)) * 128, 0, 0);
            __builtin_prefetch(V  + ((size_t)h * N + kbase + 64 + (tid >> 2)) * 128, 0, 0);
        }
        __syncthreads();

        // S = Q K^T : per n-tile, batch 4 B-frag loads then 4 chained WMMAs
        v8f s[4];
#pragma unroll
        for (int j = 0; j < 4; ++j) {
            v16h bk[4];
#pragma unroll
            for (int d = 0; d < 4; ++d) bk[d] = frag_b16(sK + j * 16 * 128 + d * 32, 128);
            v8f a = {};
#pragma unroll
            for (int d = 0; d < 4; ++d) a = wmma_f16(qf[d], bk[d], a);
            s[j] = a;
        }

        // scale + causal mask + online softmax stats (rows live per half-wave)
        float newm[8];
#pragma unroll
        for (int r = 0; r < 8; ++r) {
            int mrow = qrow0 + mh + r;
            float tmax = -3.0e38f;
#pragma unroll
            for (int j = 0; j < 4; ++j) {
                int key = kbase + j * 16 + nl;
                float x = s[j][r] * scale;
                x = (key <= mrow) ? x : -1.0e20f;
                s[j][r] = x;
                tmax = fmaxf(tmax, x);
            }
#pragma unroll
            for (int off = 1; off < 16; off <<= 1)
                tmax = fmaxf(tmax, __shfl_xor(tmax, off, 16));
            newm[r] = fmaxf(mi[r], tmax);
        }

#pragma unroll
        for (int r = 0; r < 8; ++r) {
            float rsum = 0.0f;
#pragma unroll
            for (int j = 0; j < 4; ++j) {
                float p = __expf(s[j][r] - newm[r]);
                rsum += p;
                sP[wave][(mh + r) * 64 + j * 16 + nl] = (_Float16)p;
            }
#pragma unroll
            for (int off = 1; off < 16; off <<= 1)
                rsum += __shfl_xor(rsum, off, 16);
            float alpha = __expf(mi[r] - newm[r]);
            li[r] = li[r] * alpha + rsum;
            mi[r] = newm[r];
#pragma unroll
            for (int j2 = 0; j2 < 8; ++j2) o[j2][r] *= alpha;
        }

        // O += P V : P-frags are invariant across d-tiles -> hoist
        v16h pa[2];
#pragma unroll
        for (int ks = 0; ks < 2; ++ks) pa[ks] = frag_a16(&sP[wave][ks * 32], 64);
#pragma unroll
        for (int j2 = 0; j2 < 8; ++j2) {
            v16h vb0 = frag_b16(sVt + j2 * 16 * 64, 64);
            v16h vb1 = frag_b16(sVt + j2 * 16 * 64 + 32, 64);
            o[j2] = wmma_f16(pa[0], vb0, o[j2]);
            o[j2] = wmma_f16(pa[1], vb1, o[j2]);
        }
    }

    // normalize by row sums, emit f16 [N][H*128]
#pragma unroll
    for (int r = 0; r < 8; ++r) {
        float inv = 1.0f / li[r];
        int mrow = qrow0 + mh + r;
#pragma unroll
        for (int j2 = 0; j2 < 8; ++j2) {
            Oa[(size_t)mrow * (H * 128) + h * 128 + j2 * 16 + nl] =
                (_Float16)(o[j2][r] * inv);
        }
    }
}

// ---------------------------------------------------------------------------
// Host-side orchestration
// ---------------------------------------------------------------------------
extern "C" void kernel_launch(void* const* d_in, const int* in_sizes, int n_in,
                              void* d_out, int out_size, void* d_ws, size_t ws_size,
                              hipStream_t stream) {
    const int N = 4096, C = 2048, H = 16, C3 = 3 * C;
    const float* x     = (const float*)d_in[0];
    const float* Wqkv  = (const float*)d_in[1];
    const float* Wproj = (const float*)d_in[2];
    const float* bproj = (const float*)d_in[3];
    float* out = (float*)d_out;

    // workspace layout (~208 MB)
    char* ws = (char*)d_ws;
    _Float16* x_h  = (_Float16*)(ws);                           //  16 MB
    _Float16* wq_h = (_Float16*)(ws + (size_t)16  * (1 << 20)); //  24 MB
    _Float16* wp_h = (_Float16*)(ws + (size_t)40  * (1 << 20)); //   8 MB
    float*    qkv  = (float*)   (ws + (size_t)48  * (1 << 20)); //  96 MB
    _Float16* Qh   = (_Float16*)(ws + (size_t)144 * (1 << 20)); //  16 MB
    _Float16* Kh   = (_Float16*)(ws + (size_t)160 * (1 << 20)); //  16 MB
    _Float16* Vh   = (_Float16*)(ws + (size_t)176 * (1 << 20)); //  16 MB
    _Float16* Ah   = (_Float16*)(ws + (size_t)192 * (1 << 20)); //  16 MB

    // 1) downcast operands to f16
    {
        size_t nx = (size_t)N * C, nq = (size_t)C3 * C, np = (size_t)C * C;
        cast_f32_to_f16<<<dim3((nx + 255) / 256), dim3(256), 0, stream>>>(x, x_h, nx);
        cast_f32_to_f16<<<dim3((nq + 255) / 256), dim3(256), 0, stream>>>(Wqkv, wq_h, nq);
        cast_f32_to_f16<<<dim3((np + 255) / 256), dim3(256), 0, stream>>>(Wproj, wp_h, np);
    }

    // 2) qkv = x @ Wqkv^T   [4096 x 6144] f32
    gemm_f16_wmma<false><<<dim3(C3 / 128, N / 128), dim3(256), 0, stream>>>(
        x_h, wq_h, nullptr, qkv, N, C3, C);

    // 3) RoPE + repack to [H][N][128] f16
    {
        size_t nt = (size_t)N * H * 64;
        rope_split_kernel<<<dim3((nt + 255) / 256), dim3(256), 0, stream>>>(
            qkv, Qh, Kh, Vh, N, H);
    }

    // 4) causal flash attention -> Ah [N][C] f16
    flash_attn_kernel<<<dim3(N / 128, H), dim3(256), 0, stream>>>(Qh, Kh, Vh, Ah, N, H);

    // 5) out = Ah @ Wproj^T + b   [4096 x 2048] f32
    gemm_f16_wmma<true><<<dim3(C / 128, N / 128), dim3(256), 0, stream>>>(
        Ah, wp_h, bproj, out, N, C, C);
}
